// SelfAttention_10746008175113
// MI455X (gfx1250) — matrix operand order repaired and checked by reading, not compiled, not run on hardware
//
#include <hip/hip_runtime.h>

#define DIM   256
#define NH    4
#define HD    64
#define HWSZ  1024
#define BATCH 16
#define SCALE 0.125f  /* HD^-0.5 */

typedef _Float16 h16;
typedef __attribute__((ext_vector_type(16))) _Float16 v16h;
typedef __attribute__((ext_vector_type(8)))  _Float16 v8h;
typedef __attribute__((ext_vector_type(8)))  float    v8f;
typedef __attribute__((ext_vector_type(4)))  float    v4f;
typedef __attribute__((ext_vector_type(4)))  int      i32x4;

// ---------------------------------------------------------------- async LDS
#if defined(__gfx1250__) && __has_builtin(__builtin_amdgcn_global_load_async_to_lds_b128)
#define HAS_ASYNC_LDS 1
typedef __attribute__((address_space(1))) i32x4 as1_i32x4;
typedef __attribute__((address_space(3))) i32x4 as3_i32x4;
#if __has_builtin(__builtin_amdgcn_s_wait_asynccnt)
#define WAIT_ASYNC(n) __builtin_amdgcn_s_wait_asynccnt(n)
#else
#define WAIT_ASYNC(n) asm volatile("s_wait_asynccnt %0" ::"i"(n) : "memory")
#endif
#define ASYNC_B128(g, l, off) \
  __builtin_amdgcn_global_load_async_to_lds_b128((as1_i32x4*)(g), (as3_i32x4*)(l), (off), 0)
#else
#define HAS_ASYNC_LDS 0
#define WAIT_ASYNC(n) do { } while (0)
#endif

// copy 32 bytes global->LDS (2 x b128)
__device__ __forceinline__ void cp32(const h16* g, h16* l) {
#if HAS_ASYNC_LDS
  ASYNC_B128(g, l, 0);
  ASYNC_B128(g, l, 16);
#else
  ((v8h*)l)[0] = ((const v8h*)g)[0];
  ((v8h*)l)[1] = ((const v8h*)g)[1];
#endif
}
// copy 64 bytes global->LDS (4 x b128)
__device__ __forceinline__ void cp64(const h16* g, h16* l) {
#if HAS_ASYNC_LDS
  ASYNC_B128(g, l, 0);
  ASYNC_B128(g, l, 16);
  ASYNC_B128(g, l, 32);
  ASYNC_B128(g, l, 48);
#else
  ((v8h*)l)[0] = ((const v8h*)g)[0];
  ((v8h*)l)[1] = ((const v8h*)g)[1];
  ((v8h*)l)[2] = ((const v8h*)g)[2];
  ((v8h*)l)[3] = ((const v8h*)g)[3];
#endif
}

// ---------------------------------------------------------------- WMMA utils
__device__ __forceinline__ v8f wmma32(v16h a, v16h b, v8f c) {
  return __builtin_amdgcn_wmma_f32_16x16x32_f16(false, a, false, b,
                                                (short)0, c, false, false);
}

// A fragment (16x32 f16): lane m=l&15, half=l>>4 holds K runs
// [half*8, half*8+8) and [16+half*8, 16+half*8+8)   (ISA 7.12.2)
__device__ __forceinline__ v16h load_a_frag(const h16* base, int ld, int k0, int lane) {
  const int m = lane & 15, hf = lane >> 4;
  const h16* p = base + m * ld + k0 + hf * 8;
  union { v16h v; v8h h[2]; } u;
  u.h[0] = *(const v8h*)(p);
  u.h[1] = *(const v8h*)(p + 16);
  return u.v;
}

// B fragment (32x16 f16) read from transposed tile B_T[n][k]:
// lane n=l&15, half=l>>4 holds K runs [half*8, half*8+8) and [16+half*8, 16+half*8+8)
__device__ __forceinline__ v16h load_b_frag(const h16* baseT, int ld, int k0, int lane) {
  const int n = lane & 15, hf = lane >> 4;
  const h16* p = baseT + n * ld + k0 + hf * 8;
  union { v16h v; v8h h[2]; } u;
  u.h[0] = *(const v8h*)(p);
  u.h[1] = *(const v8h*)(p + 16);
  return u.v;
}

// ---------------------------------------------------------------- kernel 0
// one-shot transpose+downconvert: xT[b][p][c] (f16) = x[b][c][p] (f32)
// grid (16, 4, 16) : p-tiles(64) x c-tiles(64) x batch
__global__ void __launch_bounds__(256)
transpose_kernel(const float* __restrict__ x, h16* __restrict__ xT) {
  const int ldL = 72;
  __shared__ h16 Ls[64 * 72];
  const int b = blockIdx.z;
  const int p0 = blockIdx.x * 64, c0 = blockIdx.y * 64;
  const int t = threadIdx.x;
  {
    const int cl = t >> 2, pc = (t & 3) * 16;
    const float* src = x + ((size_t)b * DIM + c0 + cl) * HWSZ + p0 + pc;
    h16* dst = Ls + cl * ldL + pc;
    #pragma unroll
    for (int u = 0; u < 16; u += 4) {
      v4f f = *(const v4f*)(src + u);
      dst[u+0] = (h16)f.x; dst[u+1] = (h16)f.y;
      dst[u+2] = (h16)f.z; dst[u+3] = (h16)f.w;
    }
  }
  __syncthreads();
  {
    const int pl = t >> 2, cc = (t & 3) * 16;
    union { v16h v; h16 e[16]; } u;
    #pragma unroll
    for (int j = 0; j < 16; ++j) u.e[j] = Ls[(cc + j) * ldL + pl];
    *(v16h*)(xT + ((size_t)b * HWSZ + p0 + pl) * DIM + c0 + cc) = u.v;
  }
}

// ---------------------------------------------------------------- kernel 1
// qkv = W x + b ; scatter Qrow[b,h,q,d]*SCALE, Krow[b,h,k,d], Vcol[b,h,d,k]
// grid (6, 8, 16)
__global__ void __launch_bounds__(256)
qkv_kernel(const h16* __restrict__ xT, const float* __restrict__ qkv_w,
           const float* __restrict__ qkv_b,
           h16* __restrict__ Qrow, h16* __restrict__ Krow, h16* __restrict__ Vcol) {
  const int ldA = 48, ldB = 48;
  __shared__ __align__(64) h16 As[128 * 48];   // w tile   [m=o][k=c]
  __shared__ __align__(64) h16 Bs[128 * 48];   // xT tile  [n=p][k=c]
  const int b = blockIdx.z;
  const int Mbase = blockIdx.x * 128, Nbase = blockIdx.y * 128;
  const int t = threadIdx.x, lane = t & 31, wave = t >> 5;
  const int wm = wave >> 1, wn = wave & 1;     // waves: 4(M) x 2(N)

  v8f acc[2][4];
  #pragma unroll
  for (int i = 0; i < 2; ++i)
    #pragma unroll
    for (int j = 0; j < 4; ++j) acc[i][j] = {};

  for (int k0 = 0; k0 < DIM; k0 += 32) {
    __syncthreads();
    { // B: pure f16 copy (async), 32B/thread
      const int row = t >> 1, cc = (t & 1) * 16;
      cp32(xT + ((size_t)b * HWSZ + Nbase + row) * DIM + k0 + cc,
           Bs + row * ldB + cc);
    }
    { // A: f32 -> f16 convert copy (overlaps the async B transfer)
      const int row = t >> 1, cc = (t & 1) * 16;
      const float* src = qkv_w + (size_t)(Mbase + row) * DIM + k0 + cc;
      h16* dst = As + row * ldA + cc;
      #pragma unroll
      for (int u = 0; u < 16; u += 4) {
        v4f f = *(const v4f*)(src + u);
        dst[u+0] = (h16)f.x; dst[u+1] = (h16)f.y;
        dst[u+2] = (h16)f.z; dst[u+3] = (h16)f.w;
      }
    }
    WAIT_ASYNC(0);
    __syncthreads();
    v16h af[2];
    #pragma unroll
    for (int mt = 0; mt < 2; ++mt)
      af[mt] = load_a_frag(As + (wm*32 + mt*16)*ldA, ldA, 0, lane);
    #pragma unroll
    for (int nt = 0; nt < 4; ++nt) {
      v16h bf = load_b_frag(Bs + (wn*64 + nt*16)*ldB, ldB, 0, lane);
      #pragma unroll
      for (int mt = 0; mt < 2; ++mt)
        acc[mt][nt] = wmma32(af[mt], bf, acc[mt][nt]);
    }
  }

  const int hf = lane >> 4, nn = lane & 15;   // C layout: m=r+8*hf, n=nn
  #pragma unroll
  for (int mt = 0; mt < 2; ++mt)
    #pragma unroll
    for (int nt = 0; nt < 4; ++nt)
      #pragma unroll
      for (int r = 0; r < 8; ++r) {
        const int o = Mbase + wm*32 + mt*16 + r + 8*hf;
        const int p = Nbase + wn*64 + nt*16 + nn;
        const float val = acc[mt][nt][r] + qkv_b[o];
        if (o < DIM) {
          const int h = o >> 6, d = o & 63;
          Qrow[(((size_t)b*NH + h)*HWSZ + p)*HD + d] = (h16)(val * SCALE);
        } else if (o < 2*DIM) {
          const int oo = o - DIM, h = oo >> 6, d = oo & 63;
          Krow[(((size_t)b*NH + h)*HWSZ + p)*HD + d] = (h16)val;
        } else {
          const int oo = o - 2*DIM, h = oo >> 6, d = oo & 63;
          Vcol[(((size_t)b*NH + h)*HD + d)*HWSZ + p] = (h16)val;
        }
      }
}

// ---------------------------------------------------------------- kernel 2
// Fused flash attention, async double-buffered K/V tiles. grid (8, NH, 16)
#define KS_N (128 * 80)
#define VS_N (64 * 144)
__global__ void __launch_bounds__(256)
attn_kernel(const h16* __restrict__ Qrow, const h16* __restrict__ Krow,
            const h16* __restrict__ Vcol, h16* __restrict__ Orow) {
  const int ldK = 80, ldV = 144, ldP = 144;
  __shared__ __align__(64) h16 Ks[2 * KS_N];
  __shared__ __align__(64) h16 Vs[2 * VS_N];
  __shared__ __align__(64) h16 Ps[8][16 * 144];
  const int b = blockIdx.z, h = blockIdx.y;
  const int Qbase = blockIdx.x * 128;
  const int t = threadIdx.x, lane = t & 31, wave = t >> 5;
  const size_t bh = (size_t)b * NH + h;
  const h16* Qp = Qrow + bh * HWSZ * HD;
  const h16* Kp = Krow + bh * HWSZ * HD;
  const h16* Vp = Vcol + bh * HD * HWSZ;

  // per-thread staging coords (8 x 16B async ops per tile)
  const int rowK = t >> 1, cK = (t & 1) * 32;
  const int dV   = t >> 2, cV = (t & 3) * 32;

  v16h aq[2];                                   // resident Q fragments
  #pragma unroll
  for (int ks = 0; ks < 2; ++ks)
    aq[ks] = load_a_frag(Qp + (size_t)(Qbase + wave*16) * HD, HD, ks*32, lane);

  v8f accO[4];
  #pragma unroll
  for (int i = 0; i < 4; ++i) accO[i] = {};
  float mrow[8], lrow[8];
  #pragma unroll
  for (int r = 0; r < 8; ++r) { mrow[r] = -__builtin_inff(); lrow[r] = 0.f; }

  const int hf = lane >> 4, nn = lane & 15;

  // prologue: stage tile 0 into buffer 0
  cp64(Kp + (size_t)rowK * HD + cK, Ks + rowK * ldK + cK);
  cp64(Vp + (size_t)dV * HWSZ + cV, Vs + dV * ldV + cV);

  int cb = 0;
  for (int Kb = 0; Kb < HWSZ; Kb += 128) {
    __syncthreads();               // prefetch-target buffer free of readers
    if (Kb + 128 < HWSZ) {         // stage next tile into other buffer
      const int nb = cb ^ 1;
      cp64(Kp + (size_t)(Kb + 128 + rowK) * HD + cK, Ks + nb*KS_N + rowK*ldK + cK);
      cp64(Vp + (size_t)dV * HWSZ + (Kb + 128) + cV, Vs + nb*VS_N + dV*ldV + cV);
      WAIT_ASYNC(8);               // 8 oldest ops = current tile complete
    } else {
      WAIT_ASYNC(0);
    }
    __syncthreads();               // current tile visible to all waves
    const h16* KsC = Ks + cb * KS_N;
    const h16* VsC = Vs + cb * VS_N;

    // S strip = Q K^T : 16 x 128 per wave (scale pre-folded into Q)
    v8f s[8];
    #pragma unroll
    for (int nt = 0; nt < 8; ++nt) {
      v16h b0 = load_b_frag(KsC + nt*16*ldK, ldK, 0,  lane);
      v16h b1 = load_b_frag(KsC + nt*16*ldK, ldK, 32, lane);
      v8f z = {};
      z = wmma32(aq[0], b0, z);
      s[nt] = wmma32(aq[1], b1, z);
    }

    // online softmax (row m = r + 8*hf; reduce across 16 lanes of a half)
    float scl[8];
    #pragma unroll
    for (int r = 0; r < 8; ++r) {
      float tmax = s[0][r];
      #pragma unroll
      for (int nt = 1; nt < 8; ++nt) tmax = fmaxf(tmax, s[nt][r]);
      #pragma unroll
      for (int msk = 1; msk < 16; msk <<= 1)
        tmax = fmaxf(tmax, __shfl_xor(tmax, msk, 32));
      const float mnew = fmaxf(mrow[r], tmax);
      scl[r] = __expf(mrow[r] - mnew);
      mrow[r] = mnew;
    }
    #pragma unroll
    for (int r = 0; r < 8; ++r) {
      float rs = 0.f;
      #pragma unroll
      for (int nt = 0; nt < 8; ++nt) {
        const float pv = __expf(s[nt][r] - mrow[r]);
        rs += pv;
        Ps[wave][(r + 8*hf) * ldP + nt*16 + nn] = (h16)pv;
      }
      #pragma unroll
      for (int msk = 1; msk < 16; msk <<= 1) rs += __shfl_xor(rs, msk, 32);
      lrow[r] = lrow[r] * scl[r] + rs;
    }
    #pragma unroll
    for (int dt = 0; dt < 4; ++dt)
      #pragma unroll
      for (int r = 0; r < 8; ++r) accO[dt][r] *= scl[r];

    __syncthreads();               // P strip C-layout -> A-layout round trip

    // O += P V^T
    v16h ap[4];
    #pragma unroll
    for (int ks = 0; ks < 4; ++ks)
      ap[ks] = load_a_frag(Ps[wave], ldP, ks*32, lane);
    #pragma unroll
    for (int dt = 0; dt < 4; ++dt)
      #pragma unroll
      for (int ks = 0; ks < 4; ++ks) {
        v16h bv = load_b_frag(VsC + dt*16*ldV, ldV, ks*32, lane);
        accO[dt] = wmma32(ap[ks], bv, accO[dt]);
      }
    cb ^= 1;
  }

  // normalize + store Orow[b][q][c] (ready-made B_T tile for proj)
  #pragma unroll
  for (int r = 0; r < 8; ++r) lrow[r] = 1.f / lrow[r];
  h16* Op = Orow + ((size_t)b * HWSZ + Qbase + wave*16) * DIM + h * HD;
  #pragma unroll
  for (int dt = 0; dt < 4; ++dt)
    #pragma unroll
    for (int r = 0; r < 8; ++r) {
      const int m = r + 8*hf, d = dt*16 + nn;
      Op[(size_t)m * DIM + d] = (h16)(accO[dt][r] * lrow[r]);
    }
}

// ---------------------------------------------------------------- kernel 3
// out = proj_w @ O^T + proj_b + x ; grid (2, 8, 16)
__global__ void __launch_bounds__(256)
proj_kernel(const h16* __restrict__ Orow, const float* __restrict__ proj_w,
            const float* __restrict__ proj_b, const float* __restrict__ x,
            float* __restrict__ out) {
  const int ldA = 48, ldB = 48;
  __shared__ __align__(64) h16 As[128 * 48];
  __shared__ __align__(64) h16 Bs[128 * 48];
  const int b = blockIdx.z;
  const int Mbase = blockIdx.x * 128, Nbase = blockIdx.y * 128;
  const int t = threadIdx.x, lane = t & 31, wave = t >> 5;
  const int wm = wave >> 1, wn = wave & 1;

  v8f acc[2][4];
  #pragma unroll
  for (int i = 0; i < 2; ++i)
    #pragma unroll
    for (int j = 0; j < 4; ++j) acc[i][j] = {};

  for (int k0 = 0; k0 < DIM; k0 += 32) {
    __syncthreads();
    { // B: pure f16 copy (async)
      const int row = t >> 1, cc = (t & 1) * 16;
      cp32(Orow + ((size_t)b * HWSZ + Nbase + row) * DIM + k0 + cc,
           Bs + row * ldB + cc);
    }
    { // A: proj_w f32 -> f16
      const int row = t >> 1, cc = (t & 1) * 16;
      const float* src = proj_w + (size_t)(Mbase + row) * DIM + k0 + cc;
      h16* dst = As + row * ldA + cc;
      #pragma unroll
      for (int u = 0; u < 16; u += 4) {
        v4f f = *(const v4f*)(src + u);
        dst[u+0] = (h16)f.x; dst[u+1] = (h16)f.y;
        dst[u+2] = (h16)f.z; dst[u+3] = (h16)f.w;
      }
    }
    WAIT_ASYNC(0);
    __syncthreads();
    v16h af[2];
    #pragma unroll
    for (int mt = 0; mt < 2; ++mt)
      af[mt] = load_a_frag(As + (wm*32 + mt*16)*ldA, ldA, 0, lane);
    #pragma unroll
    for (int nt = 0; nt < 4; ++nt) {
      v16h bf = load_b_frag(Bs + (wn*64 + nt*16)*ldB, ldB, 0, lane);
      #pragma unroll
      for (int mt = 0; mt < 2; ++mt)
        acc[mt][nt] = wmma32(af[mt], bf, acc[mt][nt]);
    }
  }

  const int hf = lane >> 4, nn = lane & 15;
  #pragma unroll
  for (int mt = 0; mt < 2; ++mt)
    #pragma unroll
    for (int nt = 0; nt < 4; ++nt)
      #pragma unroll
      for (int r = 0; r < 8; ++r) {
        const int o = Mbase + wm*32 + mt*16 + r + 8*hf;
        const int p = Nbase + wn*64 + nt*16 + nn;
        const size_t idx = ((size_t)b * DIM + o) * HWSZ + p;
        out[idx] = acc[mt][nt][r] + proj_b[o] + x[idx];
      }
}

// ---------------------------------------------------------------- launch
extern "C" void kernel_launch(void* const* d_in, const int* in_sizes, int n_in,
                              void* d_out, int out_size, void* d_ws, size_t ws_size,
                              hipStream_t stream) {
  (void)in_sizes; (void)n_in; (void)out_size; (void)ws_size;
  const float* x      = (const float*)d_in[0];
  const float* qkv_w  = (const float*)d_in[1];
  const float* qkv_b  = (const float*)d_in[2];
  const float* proj_w = (const float*)d_in[3];
  const float* proj_b = (const float*)d_in[4];
  float* out = (float*)d_out;

  char* ws = (char*)d_ws;
  const size_t seg = (size_t)BATCH * NH * HWSZ * HD * sizeof(h16);  // 8 MB
  h16* Qrow = (h16*)(ws);
  h16* Krow = (h16*)(ws + seg);
  h16* Vcol = (h16*)(ws + 2*seg);
  h16* Orow = (h16*)(ws + 3*seg);
  h16* xT   = (h16*)(ws + 4*seg);

  transpose_kernel<<<dim3(16, 4, BATCH), 256, 0, stream>>>(x, xT);
  qkv_kernel<<<dim3(6, 8, BATCH), 256, 0, stream>>>(xT, qkv_w, qkv_b, Qrow, Krow, Vcol);
  attn_kernel<<<dim3(HWSZ/128, NH, BATCH), 256, 0, stream>>>(Qrow, Krow, Vcol, Orow);
  proj_kernel<<<dim3(2, 8, BATCH), 256, 0, stream>>>(Orow, proj_w, proj_b, x, out);
}
